// HierarchicalGATLayer_8521215115938
// MI455X (gfx1250) — matrix operand order repaired and checked by reading, not compiled, not run on hardware
//
#include <hip/hip_runtime.h>
#include <math.h>

#define N_NODES 20000
#define IN_CH   256
#define HID     64
#define HEADS   4
#define OUT_CH  256
#define NEDGE   320000
#define ETOT    (NEDGE + N_NODES)   // edges + self loops
#define FEAT    (HEADS * HID)       // 256
#define CATW    (2 * FEAT)          // 512

typedef __attribute__((ext_vector_type(16))) __bf16       v16bf;
typedef __attribute__((ext_vector_type(8)))  float        v8f;
typedef __attribute__((ext_vector_type(4)))  unsigned int v4u;

union Frag { v16bf v; v4u u[2]; };

__device__ __forceinline__ unsigned short f32_to_bf16(float f) {
    unsigned int u = __float_as_uint(f);
    u += 0x7FFFu + ((u >> 16) & 1u);   // round-to-nearest-even
    return (unsigned short)(u >> 16);
}

// float atomic max via signed/unsigned int atomics (IEEE order trick)
__device__ __forceinline__ void atomicMaxF(float* addr, float val) {
    if (val >= 0.0f) atomicMax((int*)addr, __float_as_int(val));
    else             atomicMin((unsigned int*)addr, __float_as_uint(val));
}

// ---------------- prep kernels ----------------

__global__ void cvt_bf16_kernel(const float* __restrict__ in,
                                unsigned short* __restrict__ out, int n) {
    int t = blockIdx.x * blockDim.x + threadIdx.x;
    if (t < n) out[t] = f32_to_bf16(in[t]);
}

// out[col*K + k] = bf16(in[k*ncols + col])  (transpose to column-major for B frags)
__global__ void cvt_transpose_kernel(const float* __restrict__ in,
                                     unsigned short* __restrict__ out,
                                     int K, int ncols) {
    int t = blockIdx.x * blockDim.x + threadIdx.x;
    if (t >= K * ncols) return;
    int col = t / K, k = t - col * K;
    out[(size_t)col * K + k] = f32_to_bf16(in[(size_t)k * ncols + col]);
}

__global__ void zero_f32_kernel(float* __restrict__ p, int n) {
    int t = blockIdx.x * blockDim.x + threadIdx.x;
    if (t < n) p[t] = 0.0f;
}

__global__ void init_seg_kernel(float* __restrict__ segmax,
                                float* __restrict__ denom, int n) {
    int t = blockIdx.x * blockDim.x + threadIdx.x;
    if (t < n) { segmax[t] = -INFINITY; denom[t] = 0.0f; }
}

// ---------------- WMMA GEMM: C[M,Nc] = A_bf16[M,K] * Bt_bf16[Nc,K] ----------------
// mode 0: plain fp32 store.  mode 1: +bias then ELU.
// Each wave computes a 16x64 strip of C (4 wmma accumulators sharing one A frag).
__global__ void gemm_bf16_wmma_kernel(const unsigned short* __restrict__ A,
                                      const unsigned short* __restrict__ Bt,
                                      float* __restrict__ C,
                                      const float* __restrict__ bias,
                                      int M, int K, int Nc, int mode) {
    int wid  = (blockIdx.x * blockDim.x + threadIdx.x) >> 5;  // wave32
    int lane = threadIdx.x & 31;
    int colGroups = Nc >> 6;          // 64-wide column strips
    int rowTiles  = M >> 4;
    if (wid >= rowTiles * colGroups) return;   // wave-uniform: EXEC stays all-1s
    int rt = wid / colGroups;
    int cg = wid - rt * colGroups;
    int m0 = rt << 4, n0 = cg << 6;
    int lm = lane & 15, half = lane >> 4;

    v8f acc[4] = {v8f{}, v8f{}, v8f{}, v8f{}};

    // A frag: lane holds row m0+lm; K pairs at half*8 .. +7 and +16 .. +23 (ISA layout)
    const unsigned short* arow = A + (size_t)(m0 + lm) * K + half * 8;

    for (int k0 = 0; k0 < K; k0 += 32) {
        Frag a;
        a.u[0] = *(const v4u*)(arow + k0);        // K = k0 + half*8 + [0..7]
        a.u[1] = *(const v4u*)(arow + k0 + 16);   // K = k0 + 16 + half*8 + [0..7]
        __builtin_prefetch(arow + k0 + 32, 0, 1); // next K slab into near cache
#pragma unroll
        for (int t = 0; t < 4; ++t) {
            // B frag: lane holds column n0+t*16+lm; K at k0+half*8 .. +7 and +16 .. +23
            const unsigned short* bcol =
                Bt + (size_t)(n0 + t * 16 + lm) * K + k0 + half * 8;
            Frag b;
            b.u[0] = *(const v4u*)(bcol);
            b.u[1] = *(const v4u*)(bcol + 16);
            acc[t] = __builtin_amdgcn_wmma_f32_16x16x32_bf16(
                false, a.v, false, b.v, (short)0, acc[t], false, false);
        }
    }

#pragma unroll
    for (int t = 0; t < 4; ++t) {
        int col = n0 + t * 16 + lm;
        float badd = (mode == 1) ? bias[col] : 0.0f;
#pragma unroll
        for (int r = 0; r < 8; ++r) {
            float v = acc[t][r] + badd;                     // C/D: VGPR r -> M = r + 8*half
            if (mode == 1) v = (v > 0.0f) ? v : (__expf(v) - 1.0f);   // ELU
            C[(size_t)(m0 + r + half * 8) * Nc + col] = v;
        }
    }
}

// ---------------- attention dot products: a_src/a_dst [N,H] ----------------
// one wave per (node, head): 64 channels -> 2 per lane, shfl_xor reduction (wave32)
__global__ void att_dots_kernel(const float* __restrict__ h,
                                const float* __restrict__ att_src,
                                const float* __restrict__ att_dst,
                                float* __restrict__ a_src,
                                float* __restrict__ a_dst) {
    int wid  = (blockIdx.x * blockDim.x + threadIdx.x) >> 5;
    int lane = threadIdx.x & 31;
    if (wid >= N_NODES * HEADS) return;
    int n = wid >> 2, head = wid & 3;
    const float* hr = h + (size_t)n * FEAT + head * HID;
    float h0 = hr[lane], h1 = hr[lane + 32];
    float ps = h0 * att_src[head * HID + lane] + h1 * att_src[head * HID + lane + 32];
    float pd = h0 * att_dst[head * HID + lane] + h1 * att_dst[head * HID + lane + 32];
#pragma unroll
    for (int m = 16; m >= 1; m >>= 1) {
        ps += __shfl_xor(ps, m, 32);
        pd += __shfl_xor(pd, m, 32);
    }
    if (lane == 0) { a_src[wid] = ps; a_dst[wid] = pd; }
}

// ---------------- edge softmax passes ----------------
__global__ void edge_max_kernel(const int* __restrict__ ei,
                                const float* __restrict__ a_src,
                                const float* __restrict__ a_dst,
                                float* __restrict__ e_buf,
                                float* __restrict__ segmax) {
    int t = blockIdx.x * blockDim.x + threadIdx.x;
    if (t >= ETOT * HEADS) return;
    int e = t >> 2, head = t & 3;
    int src, dst;
    if (e < NEDGE) { src = ei[e]; dst = ei[NEDGE + e]; }
    else           { src = dst = e - NEDGE; }            // self loop
    float v = a_src[src * HEADS + head] + a_dst[dst * HEADS + head];
    v = (v > 0.0f) ? v : 0.2f * v;                       // leaky_relu(0.2)
    e_buf[t] = v;
    atomicMaxF(&segmax[dst * HEADS + head], v);
}

__global__ void edge_exp_kernel(const int* __restrict__ ei,
                                float* __restrict__ e_buf,
                                const float* __restrict__ segmax,
                                float* __restrict__ denom) {
    int t = blockIdx.x * blockDim.x + threadIdx.x;
    if (t >= ETOT * HEADS) return;
    int e = t >> 2, head = t & 3;
    int dst = (e < NEDGE) ? ei[NEDGE + e] : (e - NEDGE);
    float ex = __expf(e_buf[t] - segmax[dst * HEADS + head]);
    e_buf[t] = ex;
    atomicAdd(&denom[dst * HEADS + head], ex);
}

// one block (8 waves) per edge; thread j covers feature channel j (head = j/64)
__global__ void edge_scatter_kernel(const int* __restrict__ ei,
                                    const float* __restrict__ e_buf,
                                    const float* __restrict__ denom,
                                    const float* __restrict__ h,
                                    float* __restrict__ out) {
    int e = blockIdx.x;
    int j = threadIdx.x;           // 0..255
    int head = j >> 6;
    int src, dst;
    if (e < NEDGE) { src = ei[e]; dst = ei[NEDGE + e]; }
    else           { src = dst = e - NEDGE; }
    float alpha = e_buf[e * HEADS + head] / denom[dst * HEADS + head];
    atomicAdd(&out[(size_t)dst * FEAT + j], h[(size_t)src * FEAT + j] * alpha);
}

// ---------------- concat + bias -> bf16 for combine GEMM ----------------
__global__ void combine_prep_kernel(const float* __restrict__ out_l,
                                    const float* __restrict__ out_g,
                                    const float* __restrict__ bias_l,
                                    const float* __restrict__ bias_g,
                                    unsigned short* __restrict__ cat) {
    int t = blockIdx.x * blockDim.x + threadIdx.x;
    if (t >= N_NODES * CATW) return;
    int n = t >> 9, j = t & 511;
    float v;
    if (j < FEAT) v = out_l[(size_t)n * FEAT + j] + bias_l[j];
    else          v = out_g[(size_t)n * FEAT + (j - FEAT)] + bias_g[j - FEAT];
    cat[t] = f32_to_bf16(v);
}

static inline int cdiv(long long a, int b) { return (int)((a + b - 1) / b); }

extern "C" void kernel_launch(void* const* d_in, const int* in_sizes, int n_in,
                              void* d_out, int out_size, void* d_ws, size_t ws_size,
                              hipStream_t stream) {
    const float* x         = (const float*)d_in[0];
    const int*   ei_local  = (const int*)d_in[1];
    const int*   ei_global = (const int*)d_in[2];
    const float* W_local   = (const float*)d_in[3];
    const float* as_local  = (const float*)d_in[4];
    const float* ad_local  = (const float*)d_in[5];
    const float* b_local   = (const float*)d_in[6];
    const float* W_global  = (const float*)d_in[7];
    const float* as_global = (const float*)d_in[8];
    const float* ad_global = (const float*)d_in[9];
    const float* b_global  = (const float*)d_in[10];
    const float* comb_W    = (const float*)d_in[11];
    const float* comb_b    = (const float*)d_in[12];
    float* out = (float*)d_out;

    // workspace carve-out (256B aligned)
    char* ws = (char*)d_ws;
    size_t off = 0;
    auto alloc = [&](size_t bytes) -> char* {
        char* p = ws + off;
        off = (off + bytes + 255) & ~(size_t)255;
        return p;
    };
    unsigned short* x_bf   = (unsigned short*)alloc((size_t)N_NODES * IN_CH * 2);
    unsigned short* wt_l   = (unsigned short*)alloc((size_t)IN_CH * FEAT * 2);
    unsigned short* wt_g   = (unsigned short*)alloc((size_t)IN_CH * FEAT * 2);
    unsigned short* wt_c   = (unsigned short*)alloc((size_t)CATW * OUT_CH * 2);
    float* h      = (float*)alloc((size_t)N_NODES * FEAT * 4);
    float* a_src  = (float*)alloc((size_t)N_NODES * HEADS * 4);
    float* a_dst  = (float*)alloc((size_t)N_NODES * HEADS * 4);
    float* segmax = (float*)alloc((size_t)N_NODES * HEADS * 4);
    float* denom  = (float*)alloc((size_t)N_NODES * HEADS * 4);
    float* e_buf  = (float*)alloc((size_t)ETOT * HEADS * 4);
    float* out_l  = (float*)alloc((size_t)N_NODES * FEAT * 4);
    float* out_g  = (float*)alloc((size_t)N_NODES * FEAT * 4);
    unsigned short* cat_bf = (unsigned short*)alloc((size_t)N_NODES * CATW * 2);

    const int B = 256;

    // 1) precision/layout prep
    cvt_bf16_kernel<<<cdiv((long long)N_NODES * IN_CH, B), B, 0, stream>>>(x, x_bf, N_NODES * IN_CH);
    cvt_transpose_kernel<<<cdiv((long long)IN_CH * FEAT, B), B, 0, stream>>>(W_local,  wt_l, IN_CH, FEAT);
    cvt_transpose_kernel<<<cdiv((long long)IN_CH * FEAT, B), B, 0, stream>>>(W_global, wt_g, IN_CH, FEAT);
    cvt_transpose_kernel<<<cdiv((long long)CATW * OUT_CH, B), B, 0, stream>>>(comb_W,  wt_c, CATW, OUT_CH);
    zero_f32_kernel<<<cdiv((long long)N_NODES * FEAT, B), B, 0, stream>>>(out_l, N_NODES * FEAT);
    zero_f32_kernel<<<cdiv((long long)N_NODES * FEAT, B), B, 0, stream>>>(out_g, N_NODES * FEAT);

    // 2) two GAT convs (sequential, reusing h/attention scratch)
    const int gemmWaves = (N_NODES / 16) * (FEAT / 64);       // 5000 waves
    const int gemmBlocks = cdiv((long long)gemmWaves * 32, B);
    for (int c = 0; c < 2; ++c) {
        const unsigned short* wt = c ? wt_g : wt_l;
        const int*   ei = c ? ei_global : ei_local;
        const float* as = c ? as_global : as_local;
        const float* ad = c ? ad_global : ad_local;
        float* oc = c ? out_g : out_l;

        gemm_bf16_wmma_kernel<<<gemmBlocks, B, 0, stream>>>(
            x_bf, wt, h, nullptr, N_NODES, IN_CH, FEAT, 0);
        att_dots_kernel<<<cdiv((long long)N_NODES * HEADS * 32, B), B, 0, stream>>>(
            h, as, ad, a_src, a_dst);
        init_seg_kernel<<<cdiv((long long)N_NODES * HEADS, B), B, 0, stream>>>(
            segmax, denom, N_NODES * HEADS);
        edge_max_kernel<<<cdiv((long long)ETOT * HEADS, B), B, 0, stream>>>(
            ei, a_src, a_dst, e_buf, segmax);
        edge_exp_kernel<<<cdiv((long long)ETOT * HEADS, B), B, 0, stream>>>(
            ei, e_buf, segmax, denom);
        edge_scatter_kernel<<<ETOT, B, 0, stream>>>(ei, e_buf, denom, h, oc);
    }

    // 3) concat + bias -> bf16, then combine GEMM with bias+ELU epilogue
    combine_prep_kernel<<<cdiv((long long)N_NODES * CATW, B), B, 0, stream>>>(
        out_l, out_g, b_local, b_global, cat_bf);
    const int gemm2Waves = (N_NODES / 16) * (OUT_CH / 64);
    gemm_bf16_wmma_kernel<<<cdiv((long long)gemm2Waves * 32, B), B, 0, stream>>>(
        cat_bf, wt_c, out, comb_b, N_NODES, CATW, OUT_CH, 1);

    (void)in_sizes; (void)n_in; (void)out_size; (void)ws_size;
}
